// MessagePassingEdgeModule_20504173871664
// MI455X (gfx1250) — matrix operand order repaired and checked by reading, not compiled, not run on hardware
//
#include <hip/hip_runtime.h>

typedef __attribute__((ext_vector_type(16))) _Float16 v16h;
typedef __attribute__((ext_vector_type(8)))  _Float16 v8h;
typedef __attribute__((ext_vector_type(8)))  float    v8f;

#define D    128
#define KT1  12   // 384 / 32 K-steps for GEMM1
#define KT2  4    // 128 / 32 K-steps for GEMM2
#define NT   8    // 128 / 16 N-tiles

__global__ __launch_bounds__(256)
void edge_mlp_wmma_kernel(const float* __restrict__ x,
                          const float* __restrict__ edge_attr,
                          const long long* __restrict__ edge_index,
                          const float* __restrict__ W1,
                          const float* __restrict__ b1,
                          const float* __restrict__ W2,
                          const float* __restrict__ b2,
                          float* __restrict__ out,
                          int N, int E)
{
    // Weights pre-swizzled into WMMA B-operand layout:
    // tile[kt][nt], entry [lane][j] = W[kt*32 + (lane>=16?8:0) + (j&7) + (j>=8?16:0)][nt*16 + (lane&15)]
    __shared__ _Float16 lds_w1[KT1 * NT * 32 * 16];   // 98304 B
    __shared__ _Float16 lds_w2[KT2 * NT * 32 * 16];   // 32768 B
    __shared__ _Float16 lds_hs[8][16 * 32];           // 8192 B: per-wave 16x32 K-step stage
    __shared__ float    lds_b1[D];
    __shared__ float    lds_b2[D];
    // total 140,288 B  ->  two 8-wave workgroups fit per 320KB WGP

    const int tid = threadIdx.x;

    // ---- cooperative weight preload (f32 -> f16, B-tile swizzle) ----
    for (int idx = tid; idx < 3 * D * D; idx += 256) {
        int k = idx >> 7, n = idx & 127;
        int kt = k >> 5, kin = k & 31;
        int laneg = ((kin >> 3) & 1) << 4;    // 0 or 16
        int j = (kin & 7) | ((kin >> 4) << 3);
        int nt = n >> 4, nl = n & 15;
        lds_w1[(((kt * NT + nt) * 32) + (laneg + nl)) * 16 + j] = (_Float16)W1[idx];
    }
    for (int idx = tid; idx < D * D; idx += 256) {
        int k = idx >> 7, n = idx & 127;
        int kt = k >> 5, kin = k & 31;
        int laneg = ((kin >> 3) & 1) << 4;
        int j = (kin & 7) | ((kin >> 4) << 3);
        int nt = n >> 4, nl = n & 15;
        lds_w2[(((kt * NT + nt) * 32) + (laneg + nl)) * 16 + j] = (_Float16)W2[idx];
    }
    if (tid < D) { lds_b1[tid] = b1[tid]; lds_b2[tid] = b2[tid]; }
    __syncthreads();

    const int wave = tid >> 5;
    const int lane = tid & 31;
    const int nl   = lane & 15;      // column within 16-wide tile / A-row index
    const int hig  = lane >> 4;      // lane group (0: low 16 lanes, 1: high)
    const int kb8  = hig << 3;       // A-layout K base offset (0 or 8)

    const long tile   = (long)blockIdx.x * 8 + wave;
    const long nTiles = ((long)E + 15) >> 4;
    if (tile >= nTiles) return;

    // This lane supplies A-matrix row m = nl  ->  edge tile*16 + nl
    long eRow = tile * 16 + nl;
    long eC   = eRow < E ? eRow : (long)E - 1;
    const float* pSrc = x + (long)edge_index[eC] * D;            // src gather row
    const float* pDst = x + (long)edge_index[(long)E + eC] * D;  // dst gather row
    const float* pEa  = edge_attr + eC * D;

    // ================= GEMM1: agg[16,384] @ W1[384,128] =================
    v8f acc[NT] = {};
#pragma unroll
    for (int kt = 0; kt < KT1; ++kt) {
        const float* row = (kt < 4) ? pSrc : (kt < 8) ? pDst : pEa;
        const float* base = row + ((kt & 3) << 5) + kb8;
        // A-layout for 16-bit 16x32: lane holds K = kb8+{0..7} and kb8+16+{0..7}
        float4 f0 = *(const float4*)(base);
        float4 f1 = *(const float4*)(base + 4);
        float4 f2 = *(const float4*)(base + 16);
        float4 f3 = *(const float4*)(base + 20);
        v16h a;
        a[0]=(_Float16)f0.x;  a[1]=(_Float16)f0.y;  a[2]=(_Float16)f0.z;  a[3]=(_Float16)f0.w;
        a[4]=(_Float16)f1.x;  a[5]=(_Float16)f1.y;  a[6]=(_Float16)f1.z;  a[7]=(_Float16)f1.w;
        a[8]=(_Float16)f2.x;  a[9]=(_Float16)f2.y;  a[10]=(_Float16)f2.z; a[11]=(_Float16)f2.w;
        a[12]=(_Float16)f3.x; a[13]=(_Float16)f3.y; a[14]=(_Float16)f3.z; a[15]=(_Float16)f3.w;
#pragma unroll
        for (int nt = 0; nt < NT; ++nt) {
            v16h b = *(const v16h*)&lds_w1[((kt * NT + nt) * 32 + lane) * 16];
            acc[nt] = __builtin_amdgcn_wmma_f32_16x16x32_f16(
                false, a, false, b, (short)0, acc[nt], false, false);
        }
    }

    // ===== GEMM2: h[16,128] @ W2[128,128], fused with C->A re-layout =====
    // K-step kt consumes h columns [kt*32, kt*32+32) == acc[2kt], acc[2kt+1]:
    // stage those 16x32 values (bias+relu, f16) in a 1KB per-wave slab, then
    // read back in A layout. Same-wave LDS ops are in-order, so reusing the
    // slab across kt iterations needs no barrier.
    v8f acc2[NT] = {};
#pragma unroll
    for (int kt = 0; kt < KT2; ++kt) {
#pragma unroll
        for (int t = 0; t < 2; ++t) {
            int nt  = kt * 2 + t;
            int col = nt * 16 + nl;
            float bb = lds_b1[col];
            int cs = t * 16 + nl;                      // column within stage
#pragma unroll
            for (int r = 0; r < 8; ++r) {
                float v = acc[nt][r] + bb;             // C layout: row = hig*8 + r
                lds_hs[wave][(hig * 8 + r) * 32 + cs] = (_Float16)(v > 0.0f ? v : 0.0f);
            }
        }
        v8h alo = *(const v8h*)&lds_hs[wave][nl * 32 + kb8];        // K = kb8..kb8+7
        v8h ahi = *(const v8h*)&lds_hs[wave][nl * 32 + kb8 + 16];   // K = kb8+16..kb8+23
        v16h a = __builtin_shufflevector(alo, ahi,
                                         0,1,2,3,4,5,6,7,8,9,10,11,12,13,14,15);
#pragma unroll
        for (int nt = 0; nt < NT; ++nt) {
            v16h b = *(const v16h*)&lds_w2[((kt * NT + nt) * 32 + lane) * 16];
            acc2[nt] = __builtin_amdgcn_wmma_f32_16x16x32_f16(
                false, a, false, b, (short)0, acc2[nt], false, false);
        }
    }

    // bias + relu -> f32 output (lanes 0-15 / 16-31 store two 64B row segments)
#pragma unroll
    for (int nt = 0; nt < NT; ++nt) {
        int col = nt * 16 + nl;
        float bb = lds_b2[col];
#pragma unroll
        for (int r = 0; r < 8; ++r) {
            long er = tile * 16 + hig * 8 + r;
            if (er < E) {
                float v = acc2[nt][r] + bb;
                out[er * D + col] = v > 0.0f ? v : 0.0f;
            }
        }
    }
}

extern "C" void kernel_launch(void* const* d_in, const int* in_sizes, int n_in,
                              void* d_out, int out_size, void* d_ws, size_t ws_size,
                              hipStream_t stream) {
    const float*     x          = (const float*)d_in[0];
    const float*     edge_attr  = (const float*)d_in[1];
    const long long* edge_index = (const long long*)d_in[2];   // int64 [2,E]
    const float*     W1         = (const float*)d_in[3];
    const float*     b1         = (const float*)d_in[4];
    const float*     W2         = (const float*)d_in[5];
    const float*     b2         = (const float*)d_in[6];
    float*           out        = (float*)d_out;

    int N = in_sizes[0] / D;
    int E = in_sizes[1] / D;
    int tiles  = (E + 15) / 16;       // 16 edges per wave
    int blocks = (tiles + 7) / 8;     // 8 waves (256 threads) per block

    edge_mlp_wmma_kernel<<<blocks, 256, 0, stream>>>(
        x, edge_attr, edge_index, W1, b1, W2, b2, out, N, E);
}
